// ETSFormer_14276471292453
// MI455X (gfx1250) — matrix operand order repaired and checked by reading, not compiled, not run on hardware
//
#include <hip/hip_runtime.h>
#include <hip/hip_bf16.h>
#include <math.h>

// ---------------------------------------------------------------------------
// ETSFormer forward for MI455X (gfx1250), wave32 / WMMA / TDM.
//   B=16, N=2048, TF=8, D=512, H=8, DH=64, DFF=2048, NLAYERS=2, steps=256
// Big GEMMs: v_wmma_f32_16x16x32_bf16 with BOTH operands bf16 in memory;
// LDS tiles filled by the Tensor Data Mover (tensor_load_to_lds, double
// buffered, s_wait_tensorcnt). Producer kernels emit bf16 activation mirrors
// so the GEMM inner loop is pure ds_load_b128 + wmma (no conversion VALU).
// conv1d_fft(EMA weights) == causal EMA recurrence (exact); freq_attention =
// LDS radix-2 FFT + top-4 + 4-term irfft.
// ---------------------------------------------------------------------------

typedef __attribute__((ext_vector_type(16))) __bf16        v16bf;
typedef __attribute__((ext_vector_type(8)))  float         v8f;
typedef __attribute__((ext_vector_type(4)))  unsigned int  u32x4;
typedef __attribute__((ext_vector_type(8)))  int           i32x8;
typedef __attribute__((ext_vector_type(4)))  int           i32x4;

#if defined(__gfx1250__) && __has_builtin(__builtin_amdgcn_tensor_load_to_lds)
#define ETS_TDM 1
#else
#define ETS_TDM 0
#endif

static __device__ __forceinline__ unsigned short f2bf_bits(float f) {
  union { float f; unsigned u; } x; x.f = f;
  return (unsigned short)((x.u + 0x7FFFu + ((x.u >> 16) & 1u)) >> 16);
}
static __device__ __forceinline__ __bf16 bits2bf(unsigned short s) {
  union { unsigned short s; __bf16 b; } y; y.s = s; return y.b;
}
static __device__ __forceinline__ float sigmoidf_(float v) {
  return 1.f / (1.f + __expf(-v));
}

#if ETS_TDM
static __device__ __forceinline__ unsigned lds_offset(const void* p) {
  return (unsigned)(unsigned long long)
      (__attribute__((address_space(3))) const void*)p;
}
// Issue one 2D TDM tile load global->LDS. D# per CDNA5 ISA ch.8:
// group0: [1:0]=count=1, [63:32]=lds_addr, [120:64]=global_addr, [127:126]=2
// group1: [17:16]=data_size, dim0/dim1, tile0/tile1, dim0_stride
// (6-arg builtin on this toolchain: g0, g1, g2, g3, g4, cpol)
static __device__ __forceinline__ void tdm_load_2d(
    unsigned lds_off, const void* gptr, unsigned td0, unsigned td1,
    unsigned tile0, unsigned tile1, unsigned long long stride0,
    unsigned dsz_code) {
  unsigned long long ga = (unsigned long long)gptr;
  u32x4 g0;
  g0[0] = 1u;                                            // count=1, user desc
  g0[1] = lds_off;                                       // LDS byte address
  g0[2] = (unsigned)ga;                                  // global addr lo
  g0[3] = ((unsigned)(ga >> 32) & 0x01FFFFFFu) | (2u << 30); // hi + type=2
  i32x8 g1;
  g1[0] = (int)(dsz_code << 16);                         // mask=0, data_size
  g1[1] = (int)((td0 & 0xFFFFu) << 16);                  // tensor_dim0[15:0]
  g1[2] = (int)((td0 >> 16) | ((td1 & 0xFFFFu) << 16));  // dim0 hi | dim1 lo
  g1[3] = (int)((td1 >> 16) | (tile0 << 16));            // dim1 hi | tile0
  g1[4] = (int)tile1;                                    // tile1 (tile2=0)
  g1[5] = (int)(unsigned)(stride0 & 0xFFFFFFFFu);        // dim0_stride lo
  g1[6] = (int)(unsigned)(stride0 >> 32);                // dim0_stride hi
  g1[7] = 0;
  i32x4 zz4 = {0, 0, 0, 0};
  i32x8 zz8 = {0, 0, 0, 0, 0, 0, 0, 0};
  __builtin_amdgcn_tensor_load_to_lds(g0, g1, zz4, zz4, zz8, 0);
}
#endif

// ---------------------------------------------------------------------------
// f32 -> bf16 (RNE) bulk conversion (weights).
// ---------------------------------------------------------------------------
__global__ __launch_bounds__(256) void cvt_bf16_kernel(
    const float* __restrict__ src, unsigned short* __restrict__ dst, int n) {
  int i = blockIdx.x * 256 + threadIdx.x;
  if (i < n) dst[i] = f2bf_bits(src[i]);
}

// ---------------------------------------------------------------------------
// Embedding: 1D conv over TF channels, kernel 3, pad 1.
// ---------------------------------------------------------------------------
__global__ __launch_bounds__(256) void embed_kernel(
    const float* __restrict__ x, const float* __restrict__ w,
    const float* __restrict__ eb, float* __restrict__ z) {
  const int idx = blockIdx.x * 256 + threadIdx.x;
  const int d = idx & 511;
  const int t = (idx >> 9) & 2047;
  const int b = idx >> 20;
  float acc = eb[d];
  const float* wp = w + d * 24;
  const float* xp = x + ((size_t)b * 2048) * 8;
  #pragma unroll
  for (int k = 0; k < 3; ++k) {
    int tt = t + k - 1;
    if (tt < 0 || tt >= 2048) continue;
    const float* xr = xp + (size_t)tt * 8;
    #pragma unroll
    for (int tf = 0; tf < 8; ++tf) acc += xr[tf] * wp[tf * 3 + k];
  }
  z[idx] = acc;
}

// ---------------------------------------------------------------------------
// freq_attention: per (b,d) column: 2048-pt radix-2 FFT in LDS, top-4 bins,
// 4-term irfft. Writes seasonal (f32), updates z in place, and writes zb
// (bf16 mirror of updated z) for the following WMMA GEMM.
// ---------------------------------------------------------------------------
__global__ __launch_bounds__(256) void fft_seasonal_kernel(
    float* __restrict__ z, float* __restrict__ seas,
    unsigned short* __restrict__ zb) {
  __shared__ float sre[2048];
  __shared__ float sim[2048];
  __shared__ float rmx[256];
  __shared__ int   rix[256];
  __shared__ int   selk[4];
  const int tid = threadIdx.x;
  const int b = blockIdx.x >> 9;
  const int d = blockIdx.x & 511;
  const size_t colbase = ((size_t)b * 2048) * 512 + d;
  float* zp = z + colbase;
  float* sp = seas + colbase;
  unsigned short* zbp = zb + colbase;

  for (int i = tid; i < 2048; i += 256) { sre[i] = zp[(size_t)i * 512]; sim[i] = 0.f; }
  __syncthreads();
  for (int i = tid; i < 2048; i += 256) {
    int j = (int)(__brev((unsigned)i) >> 21);
    if (i < j) {
      float tr = sre[i]; sre[i] = sre[j]; sre[j] = tr;
      float ti = sim[i]; sim[i] = sim[j]; sim[j] = ti;
    }
  }
  __syncthreads();
  for (int s = 1; s <= 11; ++s) {
    const int m = 1 << s, half = m >> 1;
    const float wbase = -6.28318530717958647692f / (float)m;
    for (int q = tid; q < 1024; q += 256) {
      int j   = q & (half - 1);
      int grp = q >> (s - 1);
      int p0  = grp * m + j;
      int p1  = p0 + half;
      float sw, cw; __sincosf(wbase * (float)j, &sw, &cw);
      float xr = sre[p1], xi = sim[p1];
      float tr = cw * xr - sw * xi;
      float ti = cw * xi + sw * xr;
      float ur = sre[p0], ui = sim[p0];
      sre[p0] = ur + tr; sim[p0] = ui + ti;
      sre[p1] = ur - tr; sim[p1] = ui - ti;
    }
    __syncthreads();
  }
  for (int r = 0; r < 4; ++r) {
    float best = -1.f; int bi = -1;
    for (int k = tid; k <= 1024; k += 256) {
      bool used = false;
      for (int j = 0; j < r; ++j) used = used || (selk[j] == k);
      if (used) continue;
      float a2 = sre[k] * sre[k] + sim[k] * sim[k];
      if (a2 > best) { best = a2; bi = k; }
    }
    rmx[tid] = best; rix[tid] = bi;
    __syncthreads();
    for (int off = 128; off > 0; off >>= 1) {
      if (tid < off) {
        bool take = (rmx[tid + off] > rmx[tid]) ||
                    (rmx[tid + off] == rmx[tid] && rix[tid + off] >= 0 &&
                     (rix[tid] < 0 || rix[tid + off] < rix[tid]));
        if (take) { rmx[tid] = rmx[tid + off]; rix[tid] = rix[tid + off]; }
      }
      __syncthreads();
    }
    if (tid == 0) selk[r] = rix[0];
    __syncthreads();
  }
  const float inv = 1.f / 2048.f;
  const float w0  = 6.28318530717958647692f / 2048.f;
  for (int t = tid; t < 2048; t += 256) {
    float acc = 0.f;
    #pragma unroll
    for (int j = 0; j < 4; ++j) {
      int k = selk[j];
      float fac = (k == 0 || k == 1024) ? 1.f : 2.f;
      int ph = (k * t) & 2047;
      float sn, cs; __sincosf(w0 * (float)ph, &sn, &cs);
      acc += fac * (sre[k] * cs - sim[k] * sn);
    }
    float sv = acc * inv;
    size_t gi = (size_t)t * 512;
    float znew = zp[gi] - sv;
    sp[gi]  = sv;
    zp[gi]  = znew;
    zbp[gi] = f2bf_bits(znew);
  }
}

// ---------------------------------------------------------------------------
// WMMA GEMM: out = epi(Ab[MxK](bf16) @ Wb[KxN](bf16) + bias, R)
// block = 256 threads (8 waves); block tile 64(M) x 128(N), K-slab 32.
// LDS: sA 64x32 bf16, sB 32x128 bf16, double buffered, filled by TDM
// (wave 0 issues tensor_load_to_lds; s_wait_tensorcnt + barrier).
// Wave tile 32x32 = 2x2 v_wmma_f32_16x16x32_bf16. Inner loop has no cvt.
// Outputs: C (f32, nullable), Cb (bf16 mirror, nullable).
// epi: 0=none, 1=sigmoid, 2=R-v, 3=R+v
// ---------------------------------------------------------------------------
static __device__ __forceinline__ float epi_apply(int epi, float v,
                                                  const float* R, size_t idx) {
  if (epi == 1) return sigmoidf_(v);
  if (epi == 2) return R[idx] - v;
  if (epi == 3) return R[idx] + v;
  return v;
}

__global__ __launch_bounds__(256) void gemm_wmma_bf16(
    const unsigned short* __restrict__ Ab,
    const unsigned short* __restrict__ Wb,
    const float* __restrict__ bias, const float* __restrict__ R,
    float* __restrict__ C, unsigned short* __restrict__ Cb,
    int M, int N, int K, int epi) {
  __shared__ unsigned short sA[2][64 * 32];
  __shared__ unsigned short sB[2][32 * 128];
  const int lane = threadIdx.x & 31;
  const int wave = threadIdx.x >> 5;
  const int wm = wave >> 2;                  // 0..1
  const int wn = wave & 3;                   // 0..3
  const int mblk = blockIdx.y * 64;
  const int nblk = blockIdx.x * 128;
  const int m0 = mblk + wm * 32;
  const int n0 = nblk + wn * 32;
  const int mrow = lane & 15;
  const int khi  = lane >> 4;                // 0/1

  v8f c00 = {}, c01 = {}, c10 = {}, c11 = {};
  const int arow0 = (wm * 32 + mrow) * 32;
  const int arow1 = arow0 + 16 * 32;
  const int brow  = wn * 32 + mrow;

#if ETS_TDM
  if (wave == 0) {
    tdm_load_2d(lds_offset(&sA[0][0]), Ab + (size_t)mblk * K,
                (unsigned)K, (unsigned)M, 32u, 64u, (unsigned long long)K, 1u);
    tdm_load_2d(lds_offset(&sB[0][0]), Wb + nblk,
                (unsigned)N, (unsigned)K, 128u, 32u, (unsigned long long)N, 1u);
    __builtin_amdgcn_s_wait_tensorcnt((short)0);
  }
  __syncthreads();
  int cur = 0;
  for (int k0 = 0; k0 < K; k0 += 32) {
    const int nxt = cur ^ 1;
    const bool more = (k0 + 32) < K;
    if (wave == 0 && more) {
      tdm_load_2d(lds_offset(&sA[nxt][0]), Ab + (size_t)mblk * K + (k0 + 32),
                  (unsigned)K, (unsigned)M, 32u, 64u, (unsigned long long)K, 1u);
      tdm_load_2d(lds_offset(&sB[nxt][0]),
                  Wb + (size_t)(k0 + 32) * N + nblk,
                  (unsigned)N, (unsigned)K, 128u, 32u, (unsigned long long)N, 1u);
    }
    const unsigned short* pA = &sA[cur][0];
    const unsigned short* pB = &sB[cur][0];
    v16bf a0, a1, b0, b1;
    #pragma unroll
    for (int i = 0; i < 16; ++i) {    // A layout: k=(v&3)*2+(v>>2)*16+khi*8+h
      int k = ((((i >> 1) & 3) << 1) | (i & 1)) + ((i >> 3) << 4) + (khi << 3);
      a0[i] = bits2bf(pA[arow0 + k]);
      a1[i] = bits2bf(pA[arow1 + k]);
    }
    #pragma unroll
    for (int i = 0; i < 16; ++i) {    // B layout: lane&15 -> N, element i -> K (same k order as A)
      int k = (i & 7) + ((i >> 3) << 4) + (khi << 3);
      b0[i] = bits2bf(pB[brow + k * 128]);
      b1[i] = bits2bf(pB[brow + 16 + k * 128]);
    }
    c00 = __builtin_amdgcn_wmma_f32_16x16x32_bf16(false, a0, false, b0, (short)0, c00, false, false);
    c01 = __builtin_amdgcn_wmma_f32_16x16x32_bf16(false, a0, false, b1, (short)0, c01, false, false);
    c10 = __builtin_amdgcn_wmma_f32_16x16x32_bf16(false, a1, false, b0, (short)0, c10, false, false);
    c11 = __builtin_amdgcn_wmma_f32_16x16x32_bf16(false, a1, false, b1, (short)0, c11, false, false);
    if (wave == 0 && more) __builtin_amdgcn_s_wait_tensorcnt((short)0);
    __syncthreads();
    cur = nxt;
  }
#else
  for (int k0 = 0; k0 < K; k0 += 32) {
    for (int i = threadIdx.x; i < 64 * 32; i += 256) {
      int r = i >> 5, cc = i & 31;
      sA[0][i] = Ab[(size_t)(mblk + r) * K + k0 + cc];
    }
    for (int i = threadIdx.x; i < 32 * 128; i += 256) {
      int r = i >> 7, cc = i & 127;
      sB[0][i] = Wb[(size_t)(k0 + r) * N + nblk + cc];
    }
    __syncthreads();
    const unsigned short* pA = &sA[0][0];
    const unsigned short* pB = &sB[0][0];
    v16bf a0, a1, b0, b1;
    #pragma unroll
    for (int i = 0; i < 16; ++i) {
      int k = ((((i >> 1) & 3) << 1) | (i & 1)) + ((i >> 3) << 4) + (khi << 3);
      a0[i] = bits2bf(pA[arow0 + k]);
      a1[i] = bits2bf(pA[arow1 + k]);
    }
    #pragma unroll
    for (int i = 0; i < 16; ++i) {
      int k = (i & 7) + ((i >> 3) << 4) + (khi << 3);
      b0[i] = bits2bf(pB[brow + k * 128]);
      b1[i] = bits2bf(pB[brow + 16 + k * 128]);
    }
    c00 = __builtin_amdgcn_wmma_f32_16x16x32_bf16(false, a0, false, b0, (short)0, c00, false, false);
    c01 = __builtin_amdgcn_wmma_f32_16x16x32_bf16(false, a0, false, b1, (short)0, c01, false, false);
    c10 = __builtin_amdgcn_wmma_f32_16x16x32_bf16(false, a1, false, b0, (short)0, c10, false, false);
    c11 = __builtin_amdgcn_wmma_f32_16x16x32_bf16(false, a1, false, b1, (short)0, c11, false, false);
    __syncthreads();
  }
#endif

  // C/D f32 layout: n = lane&15, m = r + 8*(lane>>4)
  #pragma unroll
  for (int r = 0; r < 8; ++r) {
    int mA = m0 + r + (khi << 3);
    int mB = mA + 16;
    int nA = n0 + mrow;
    int nB = nA + 16;
    size_t i00 = (size_t)mA * N + nA, i01 = (size_t)mA * N + nB;
    size_t i10 = (size_t)mB * N + nA, i11 = (size_t)mB * N + nB;
    float v00 = epi_apply(epi, c00[r] + bias[nA], R, i00);
    float v01 = epi_apply(epi, c01[r] + bias[nB], R, i01);
    float v10 = epi_apply(epi, c10[r] + bias[nA], R, i10);
    float v11 = epi_apply(epi, c11[r] + bias[nB], R, i11);
    if (C) {
      C[i00] = v00; C[i01] = v01; C[i10] = v10; C[i11] = v11;
    }
    if (Cb) {
      Cb[i00] = f2bf_bits(v00); Cb[i01] = f2bf_bits(v01);
      Cb[i10] = f2bf_bits(v10); Cb[i11] = f2bf_bits(v11);
    }
  }
}

// ---------------------------------------------------------------------------
// mhesa EMA scan: per channel c of xi[B,N,512] (f32 in, bf16 out):
// dx[t]=xi[t]-xi[t-1] (xi[-1]=init), s=(1-a)s+a*dx, out=s+(1-a)^{t+1}*init
// ---------------------------------------------------------------------------
__global__ __launch_bounds__(256) void mhesa_scan_kernel(
    const float* __restrict__ xi, const float* __restrict__ init_state,
    const float* __restrict__ alpha, unsigned short* __restrict__ gb) {
  const int idx = blockIdx.x * 256 + threadIdx.x;   // 8192 channels
  const int b = idx >> 9;
  const int c = idx & 511;
  const int h = c >> 6;
  const float a  = sigmoidf_(alpha[h]);
  const float om = 1.f - a;
  const float init = init_state[c];
  const float*    xp = xi + ((size_t)b * 2048) * 512 + c;
  unsigned short* gp = gb + ((size_t)b * 2048) * 512 + c;
  float prev = init, s = 0.f, pw = 1.f;
  for (int t = 0; t < 2048; ++t) {
    float v  = xp[(size_t)t * 512];
    float dx = v - prev; prev = v;
    s  = om * s + a * dx;
    pw *= om;
    gp[(size_t)t * 512] = f2bf_bits(s + pw * init);
  }
}

// ---------------------------------------------------------------------------
// LayerNorm in place over last dim (512), one wave32 per row; also writes
// bf16 mirror zb for the following WMMA GEMM.
// ---------------------------------------------------------------------------
__global__ __launch_bounds__(256) void layernorm_kernel(
    float* __restrict__ z, unsigned short* __restrict__ zb,
    const float* __restrict__ g, const float* __restrict__ bta) {
  const int lane = threadIdx.x & 31;
  const int row  = blockIdx.x * 8 + (threadIdx.x >> 5);
  float* zp = z + (size_t)row * 512;
  unsigned short* zbp = zb + (size_t)row * 512;
  float v[16];
  float sum = 0.f;
  #pragma unroll
  for (int j = 0; j < 16; ++j) { v[j] = zp[lane + j * 32]; sum += v[j]; }
  #pragma unroll
  for (int o = 16; o > 0; o >>= 1) sum += __shfl_xor(sum, o, 32);
  const float mu = sum * (1.f / 512.f);
  float var = 0.f;
  #pragma unroll
  for (int j = 0; j < 16; ++j) { float dd = v[j] - mu; var += dd * dd; }
  #pragma unroll
  for (int o = 16; o > 0; o >>= 1) var += __shfl_xor(var, o, 32);
  const float rs = rsqrtf(var * (1.f / 512.f) + 1e-5f);
  #pragma unroll
  for (int j = 0; j < 16; ++j) {
    int col = lane + j * 32;
    float y = (v[j] - mu) * rs * g[col] + bta[col];
    zp[col]  = y;
    zbp[col] = f2bf_bits(y);
  }
}

// ---------------------------------------------------------------------------
// Seasonal projection: proj[b,t,tf] = seas[b,t,:] @ w_s[:,tf] + b_s[tf]
// ---------------------------------------------------------------------------
__global__ __launch_bounds__(256) void seas_proj_kernel(
    const float* __restrict__ seas, const float* __restrict__ w_s,
    const float* __restrict__ b_s, float* __restrict__ proj) {
  const int idx = blockIdx.x * 256 + threadIdx.x;   // B*N*TF
  const int tf = idx & 7;
  const int row = idx >> 3;
  const float* sr = seas + (size_t)row * 512;
  float acc = b_s[tf];
  #pragma unroll 4
  for (int d = 0; d < 512; ++d) acc += sr[d] * w_s[d * 8 + tf];
  proj[idx] = acc;
}

// ---------------------------------------------------------------------------
// level_step: seasonal_term + growth_term == EMA of (2x - seasonal_proj)
// ---------------------------------------------------------------------------
__global__ __launch_bounds__(128) void level_scan_kernel(
    const float* __restrict__ xin, const float* __restrict__ proj,
    const float* __restrict__ alpha, float* __restrict__ xout) {
  const int idx = threadIdx.x;                      // 128 = B*TF channels
  const float a  = sigmoidf_(alpha[0]);
  const float om = 1.f - a;
  const int b = idx >> 3, tf = idx & 7;
  const size_t base = ((size_t)b * 2048) * 8 + tf;
  float s = 0.f;
  for (int t = 0; t < 2048; ++t) {
    size_t o = base + (size_t)t * 8;
    float v = 2.f * xin[o] - proj[o];
    s = om * s + a * v;
    xout[o] = s;
  }
}

// ---------------------------------------------------------------------------
__global__ __launch_bounds__(256) void final_bcast_kernel(
    const float* __restrict__ xl, float* __restrict__ out, int steps) {
  const int idx = blockIdx.x * 256 + threadIdx.x;   // B*steps*TF
  const int tf = idx & 7;
  const int b  = idx / (steps * 8);
  out[idx] = xl[(((size_t)b * 2048) + 2047) * 8 + tf];
}

// ---------------------------------------------------------------------------
extern "C" void kernel_launch(void* const* d_in, const int* in_sizes, int n_in,
                              void* d_out, int out_size, void* d_ws, size_t ws_size,
                              hipStream_t stream) {
  (void)in_sizes; (void)n_in; (void)ws_size;
  constexpr int B = 16, N = 2048, TF = 8, D = 512, DFF = 2048, NL = 2;
  constexpr int M = B * N;                 // 32768
  const int steps = out_size / (B * TF);   // 256

  const float* x       = (const float*)d_in[0];
  const float* embed_w = (const float*)d_in[1];
  const float* embed_b = (const float*)d_in[2];
  auto L = [&](int l, int off) { return (const float*)d_in[3 + l * 19 + off]; };
  // 0 init_state, 1 alpha, 2 w_in, 3 b_in, 4 w_out, 5 b_out, 6 ln_g, 7 ln_b,
  // 8 ff.ln_g, 9 ff.ln_b, 10 w1, 11 b1, 12 w2, 13 b2, 14 lvl.alpha,
  // 15 w_s, 16 b_s (17 w_g, 18 b_g unused)

  char* p = (char*)d_ws;
  const size_t SZ_F32 = (size_t)M * D * 4;           // 64 MB
  const size_t SZ_BF  = (size_t)M * D * 2;           // 32 MB
  float*          z    = (float*)p;          p += SZ_F32;
  float*          seas = (float*)p;          p += SZ_F32;
  float*          bufA = (float*)p;          p += SZ_F32;  // xi (f32, for scan)
  unsigned short* zb   = (unsigned short*)p; p += SZ_BF;   // bf16 mirror of z
  unsigned short* gb   = (unsigned short*)p; p += SZ_BF;   // bf16 scan output
  unsigned short* hbf  = (unsigned short*)p; p += SZ_BF;   // bf16 FF hidden chunk
  unsigned short* wbf1 = (unsigned short*)p; p += (size_t)D * DFF * 2; // 2 MB
  unsigned short* wbf2 = (unsigned short*)p; p += (size_t)D * DFF * 2; // 2 MB
  float* xa   = (float*)p;                   p += (size_t)B * N * TF * 4;
  float* xb   = (float*)p;                   p += (size_t)B * N * TF * 4;
  float* proj = (float*)p;

  auto cvt = [&](const float* src, unsigned short* dst, int n) {
    cvt_bf16_kernel<<<(n + 255) / 256, 256, 0, stream>>>(src, dst, n);
  };

  // 1) embed
  embed_kernel<<<(M * D) / 256, 256, 0, stream>>>(x, embed_w, embed_b, z);

  for (int l = 0; l < NL; ++l) {
    const float* xin  = (l == 0) ? x : xa;
    float*       xout = (l == 0) ? xa : xb;

    // 2) freq_attention: seasonal, z -= seasonal (also emits zb)
    fft_seasonal_kernel<<<B * D, 256, 0, stream>>>(z, seas, zb);

    // 3) mhesa
    cvt(L(l, 2), wbf1, D * D);                          // w_in -> bf16
    gemm_wmma_bf16<<<dim3(D / 128, M / 64), 256, 0, stream>>>(
        zb, wbf1, L(l, 3), nullptr, bufA, nullptr, M, D, D, 0); // xi f32
    mhesa_scan_kernel<<<(B * D) / 256, 256, 0, stream>>>(
        bufA, L(l, 0), L(l, 1), gb);                    // EMA scan -> bf16
    cvt(L(l, 4), wbf2, D * D);                          // w_out -> bf16
    gemm_wmma_bf16<<<dim3(D / 128, M / 64), 256, 0, stream>>>(
        gb, wbf2, L(l, 5), z, z, nullptr, M, D, D, 2);  // z -= g@w_out + b_out

    // 4) z = layernorm(z) (emits zb)
    layernorm_kernel<<<M / 8, 256, 0, stream>>>(z, zb, L(l, 6), L(l, 7));

    // 5) FF: z = layernorm(z + sigmoid(z@w1+b1)@w2+b2), chunked over M
    cvt(L(l, 10), wbf1, D * DFF);                       // w1 -> bf16
    cvt(L(l, 12), wbf2, DFF * D);                       // w2 -> bf16
    constexpr int MC = 8192;
    for (int c0 = 0; c0 < M; c0 += MC) {
      float*          zc  = z  + (size_t)c0 * D;
      unsigned short* zbc = zb + (size_t)c0 * D;
      gemm_wmma_bf16<<<dim3(DFF / 128, MC / 64), 256, 0, stream>>>(
          zbc, wbf1, L(l, 11), nullptr, nullptr, hbf, MC, DFF, D, 1); // sigmoid, bf16 only
      gemm_wmma_bf16<<<dim3(D / 128, MC / 64), 256, 0, stream>>>(
          hbf, wbf2, L(l, 13), zc, zc, nullptr, MC, D, DFF, 3);       // residual add
    }
    layernorm_kernel<<<M / 8, 256, 0, stream>>>(z, zb, L(l, 8), L(l, 9));

    // 6) level_step
    seas_proj_kernel<<<(M * TF) / 256, 256, 0, stream>>>(
        seas, L(l, 15), L(l, 16), proj);
    level_scan_kernel<<<1, B * TF, 0, stream>>>(xin, proj, L(l, 14), xout);
  }

  // 7) broadcast last timestep
  final_bcast_kernel<<<(B * steps * TF) / 256, 256, 0, stream>>>(
      xb, (float*)d_out, steps);
}
